// RNN_38036230373814
// MI455X (gfx1250) — matrix operand order repaired and checked by reading, not compiled, not run on hardware
//
#include <hip/hip_runtime.h>
#include <hip/hip_bf16.h>
#include <math.h>

// ---------------------------------------------------------------------------
// RNN (3-layer tanh, B=4096, T=20, D_IN=300, H=512, C=10) on CDNA5 WMMA.
// bf16 operands, f32 accumulation via v_wmma_f32_16x16x32_bf16.
// Async global->LDS double buffering for the per-step emb tile (ASYNCcnt).
// ---------------------------------------------------------------------------

#define RNN_B   4096
#define RNN_T   20
#define RNN_DIN 300
#define RNN_H   512
#define RNN_C   10

#define MTILE   32          // batch rows per workgroup
#define HSTR    520         // LDS row stride in bf16 (260 dwords -> bank-conflict free)
#define XKPAD   320         // D_IN padded to multiple of 32
#define XSTR    328         // LDS row stride for x tile (164 dwords)

typedef __attribute__((ext_vector_type(16))) __bf16 v16bf;
typedef __attribute__((ext_vector_type(8)))  __bf16 v8bf;
typedef __attribute__((ext_vector_type(8)))  float  v8f;

#if __has_builtin(__builtin_amdgcn_tanhf)
#define TANHF(x) __builtin_amdgcn_tanhf(x)   // hardware V_TANH_F32
#else
#define TANHF(x) tanhf(x)
#endif

// ---- WMMA fragment loaders -------------------------------------------------
// A (16x32 bf16): lane L holds row (L&15). lanes 0-15: K = kt*32 + [0..15];
// lanes 16-31: K = kt*32 + [16..31] (matches B fragment order).
__device__ __forceinline__ v16bf load_afrag_lds(const __bf16* base, int stride,
                                                int row0, int kt, int lane) {
  int row = row0 + (lane & 15);
  int k0  = kt * 32 + ((lane & 16) ? 16 : 0);
  const __bf16* p = base + row * stride + k0;
  v8bf lo = *(const v8bf*)(p);        // 16B -> ds_load_b128
  v8bf hi = *(const v8bf*)(p + 8);    // 16B -> ds_load_b128
  return __builtin_shufflevector(lo, hi, 0, 1, 2, 3, 4, 5, 6, 7,
                                 8, 9, 10, 11, 12, 13, 14, 15);
}

// B (32x16 bf16): lane L holds column (L&15); lanes 0-15 K=0..15,
// lanes 16-31 K=16..31. Weights are stored TRANSPOSED ([N][Kpad]) so each
// lane reads 16 consecutive bf16 (32B, aligned) from global / L2.
__device__ __forceinline__ v16bf load_bfrag(const __bf16* WT, int kstride,
                                            int colbase, int kt, int lane) {
  int n  = colbase + (lane & 15);
  int k0 = kt * 32 + ((lane & 16) ? 16 : 0);
  return *(const v16bf*)(WT + (size_t)n * kstride + k0);
}

// ---- async global -> LDS copy of one 32x512 bf16 tile (contiguous source) --
// GLOBAL_LOAD_ASYNC_TO_LDS_B128: per-lane 16B, tracked with ASYNCcnt.
__device__ __forceinline__ void async_copy_tile(const __bf16* __restrict__ src,
                                                __bf16* __restrict__ dst,
                                                int tid) {
#pragma unroll
  for (int it = 0; it < (MTILE * RNN_H / 8) / 256; ++it) {
    int idx = tid + it * 256;
    int row = idx >> 6;                 // 64 chunks of 8 bf16 per row
    int k   = (idx & 63) * 8;
    unsigned lds_off = (unsigned)(size_t)(dst + row * HSTR + k);
    unsigned long long ga = (unsigned long long)(size_t)(src + row * RNN_H + k);
    asm volatile("global_load_async_to_lds_b128 %0, %1, off"
                 :: "v"(lds_off), "v"(ga) : "memory");
  }
}

__device__ __forceinline__ void wait_async_all() {
  asm volatile("s_wait_asynccnt 0" ::: "memory");
}

// ---- one RNN layer step: H = tanh(X @ WxT' + H @ WhT' + b) -----------------
// X, H are [MTILE][HSTR] bf16 in LDS. WxT/WhT are transposed bf16 [512][512].
// 8 waves: wave w -> m-tile (w&1), 128-col quarter (w>>1).
__device__ __forceinline__ void layer_step(const __bf16* __restrict__ X,
                                           const __bf16* __restrict__ WxT,
                                           __bf16* __restrict__ H,
                                           const __bf16* __restrict__ WhT,
                                           const float* __restrict__ bias) {
  const int tid = threadIdx.x;
  const int lane = tid & 31;
  const int w = tid >> 5;
  const int m = w & 1;
  const int colbase = (w >> 1) * 128;

  v8f acc[8] = {};

  for (int kt = 0; kt < RNN_H / 32; ++kt) {          // X @ WxT
    v16bf a = load_afrag_lds(X, HSTR, m * 16, kt, lane);
#pragma unroll
    for (int nt = 0; nt < 8; ++nt) {
      v16bf b = load_bfrag(WxT, RNN_H, colbase + nt * 16, kt, lane);
      acc[nt] = __builtin_amdgcn_wmma_f32_16x16x32_bf16(
          false, a, false, b, (short)0, acc[nt], false, false);
    }
  }
  for (int kt = 0; kt < RNN_H / 32; ++kt) {          // + H @ WhT
    v16bf a = load_afrag_lds(H, HSTR, m * 16, kt, lane);
#pragma unroll
    for (int nt = 0; nt < 8; ++nt) {
      v16bf b = load_bfrag(WhT, RNN_H, colbase + nt * 16, kt, lane);
      acc[nt] = __builtin_amdgcn_wmma_f32_16x16x32_bf16(
          false, a, false, b, (short)0, acc[nt], false, false);
    }
  }

  __syncthreads();   // everyone done READING old H before anyone writes

  const int cl = lane & 15;
  const int rbase = m * 16 + ((lane & 16) ? 8 : 0);
#pragma unroll
  for (int nt = 0; nt < 8; ++nt) {
    int col = colbase + nt * 16 + cl;
    float bv = bias[col];
#pragma unroll
    for (int r = 0; r < 8; ++r) {
      float v = TANHF(acc[nt][r] + bv);
      H[(rbase + r) * HSTR + col] = (__bf16)v;
    }
  }
  __syncthreads();
}

// ---- kernel 0: fp32 [K][N] -> bf16 transposed [N][Kpad] --------------------
__global__ void wtrans_kernel(const float* __restrict__ src,
                              __bf16* __restrict__ dst,
                              int K, int N, int Kpad) {
  int i = blockIdx.x * blockDim.x + threadIdx.x;
  if (i >= N * Kpad) return;
  int n = i / Kpad, k = i - n * Kpad;
  float v = (k < K) ? src[(size_t)k * N + n] : 0.0f;
  dst[i] = (__bf16)v;
}

// ---- kernel 1: emb = x @ W_emb + b_emb  -> bf16 [T][B][H] ------------------
__global__ __launch_bounds__(256)
void emb_kernel(const float* __restrict__ x,
                const __bf16* __restrict__ WembT,   // [512][XKPAD]
                const float* __restrict__ b_emb,
                __bf16* __restrict__ emb) {          // [T][B][512]
  __shared__ __bf16 sX[MTILE * XSTR];
  const int tid = threadIdx.x;
  const int r0 = blockIdx.x * MTILE;                 // flattened row b*T + t

  for (int idx = tid; idx < MTILE * XKPAD; idx += 256) {
    int row = idx / XKPAD, k = idx - row * XKPAD;
    float v = (k < RNN_DIN) ? x[(size_t)(r0 + row) * RNN_DIN + k] : 0.0f;
    sX[row * XSTR + k] = (__bf16)v;
  }
  __syncthreads();

  const int lane = tid & 31;
  const int w = tid >> 5;
  const int m = w & 1;
  const int colbase = (w >> 1) * 128;

  v8f acc[8] = {};
  for (int kt = 0; kt < XKPAD / 32; ++kt) {
    v16bf a = load_afrag_lds(sX, XSTR, m * 16, kt, lane);
#pragma unroll
    for (int nt = 0; nt < 8; ++nt) {
      v16bf b = load_bfrag(WembT, XKPAD, colbase + nt * 16, kt, lane);
      acc[nt] = __builtin_amdgcn_wmma_f32_16x16x32_bf16(
          false, a, false, b, (short)0, acc[nt], false, false);
    }
  }

  const int cl = lane & 15;
  const int rbase = m * 16 + ((lane & 16) ? 8 : 0);
#pragma unroll
  for (int nt = 0; nt < 8; ++nt) {
    int col = colbase + nt * 16 + cl;
    float bv = b_emb[col];
#pragma unroll
    for (int r = 0; r < 8; ++r) {
      int rf = r0 + rbase + r;               // = b*T + t
      int bb = rf / RNN_T, tt = rf - bb * RNN_T;
      emb[((size_t)tt * RNN_B + bb) * RNN_H + col] = (__bf16)(acc[nt][r] + bv);
    }
  }
}

// ---- kernel 2: the recurrence + final FC -----------------------------------
__global__ __launch_bounds__(256)
void rnn_kernel(const __bf16* __restrict__ emb,     // [T][B][512]
                const __bf16* __restrict__ wx0T, const __bf16* __restrict__ wh0T,
                const float* __restrict__ b0,
                const __bf16* __restrict__ wx1T, const __bf16* __restrict__ wh1T,
                const float* __restrict__ b1,
                const __bf16* __restrict__ wx2T, const __bf16* __restrict__ wh2T,
                const float* __restrict__ b2,
                const float* __restrict__ W_fc, const float* __restrict__ b_fc,
                float* __restrict__ out) {
  __shared__ __bf16 sE0[MTILE * HSTR];   // double-buffered emb tiles
  __shared__ __bf16 sE1[MTILE * HSTR];
  __shared__ __bf16 sH0[MTILE * HSTR];
  __shared__ __bf16 sH1[MTILE * HSTR];
  __shared__ __bf16 sH2[MTILE * HSTR];

  const int tid = threadIdx.x;
  const int brow0 = blockIdx.x * MTILE;

  for (int i = tid; i < MTILE * HSTR; i += 256) {
    sH0[i] = (__bf16)0.0f; sH1[i] = (__bf16)0.0f; sH2[i] = (__bf16)0.0f;
  }

  // kick off async load of step 0's tile while we zero/init
  async_copy_tile(emb + (size_t)brow0 * RNN_H, sE0, tid);
  __syncthreads();

#pragma unroll 1
  for (int t = 0; t < RNN_T; ++t) {
    __bf16* cur = (t & 1) ? sE1 : sE0;
    __bf16* nxt = (t & 1) ? sE0 : sE1;

    wait_async_all();      // this wave's async chunks of tile t are in LDS
    __syncthreads();       // -> every wave's chunks are in LDS

    if (t + 1 < RNN_T)     // overlap next tile's DMA with 3 layers of compute
      async_copy_tile(emb + ((size_t)(t + 1) * RNN_B + brow0) * RNN_H, nxt, tid);

    layer_step(cur, wx0T, sH0, wh0T, b0);
    layer_step(sH0, wx1T, sH1, wh1T, b1);
    layer_step(sH1, wx2T, sH2, wh2T, b2);
  }

  // final FC: out[32,10] = h2 @ W_fc + b_fc (tiny; plain VALU)
  if (tid < MTILE) {
    int row = tid;
    float acc[RNN_C];
#pragma unroll
    for (int c = 0; c < RNN_C; ++c) acc[c] = b_fc[c];
    for (int k = 0; k < RNN_H; ++k) {
      float hv = (float)sH2[row * HSTR + k];
#pragma unroll
      for (int c = 0; c < RNN_C; ++c) acc[c] += hv * W_fc[k * RNN_C + c];
    }
#pragma unroll
    for (int c = 0; c < RNN_C; ++c)
      out[(size_t)(brow0 + row) * RNN_C + c] = acc[c];
  }
}

// ---------------------------------------------------------------------------
extern "C" void kernel_launch(void* const* d_in, const int* in_sizes, int n_in,
                              void* d_out, int out_size, void* d_ws, size_t ws_size,
                              hipStream_t stream) {
  const float* x     = (const float*)d_in[0];
  const float* W_emb = (const float*)d_in[1];
  const float* b_emb = (const float*)d_in[2];
  const float* wx0   = (const float*)d_in[3];
  const float* wh0   = (const float*)d_in[4];
  const float* b0    = (const float*)d_in[5];
  const float* wx1   = (const float*)d_in[6];
  const float* wh1   = (const float*)d_in[7];
  const float* b1    = (const float*)d_in[8];
  const float* wx2   = (const float*)d_in[9];
  const float* wh2   = (const float*)d_in[10];
  const float* b2    = (const float*)d_in[11];
  const float* W_fc  = (const float*)d_in[12];
  const float* b_fc  = (const float*)d_in[13];
  float* out = (float*)d_out;

  // workspace layout (bytes)
  char* ws = (char*)d_ws;
  const size_t szWembT = (size_t)RNN_H * XKPAD * sizeof(__bf16);   // 327,680
  const size_t szW     = (size_t)RNN_H * RNN_H * sizeof(__bf16);   // 524,288
  __bf16* WembT = (__bf16*)(ws);
  __bf16* wx0T  = (__bf16*)(ws + szWembT + 0 * szW);
  __bf16* wh0T  = (__bf16*)(ws + szWembT + 1 * szW);
  __bf16* wx1T  = (__bf16*)(ws + szWembT + 2 * szW);
  __bf16* wh1T  = (__bf16*)(ws + szWembT + 3 * szW);
  __bf16* wx2T  = (__bf16*)(ws + szWembT + 4 * szW);
  __bf16* wh2T  = (__bf16*)(ws + szWembT + 5 * szW);
  __bf16* emb   = (__bf16*)(ws + szWembT + 6 * szW);               // [T][B][512] bf16

  // weight conversion / transpose (fp32 -> bf16, [K][N] -> [N][Kpad])
  {
    int n = RNN_H * XKPAD;
    wtrans_kernel<<<(n + 255) / 256, 256, 0, stream>>>(W_emb, WembT, RNN_DIN, RNN_H, XKPAD);
    n = RNN_H * RNN_H;
    int g = (n + 255) / 256;
    wtrans_kernel<<<g, 256, 0, stream>>>(wx0, wx0T, RNN_H, RNN_H, RNN_H);
    wtrans_kernel<<<g, 256, 0, stream>>>(wh0, wh0T, RNN_H, RNN_H, RNN_H);
    wtrans_kernel<<<g, 256, 0, stream>>>(wx1, wx1T, RNN_H, RNN_H, RNN_H);
    wtrans_kernel<<<g, 256, 0, stream>>>(wh1, wh1T, RNN_H, RNN_H, RNN_H);
    wtrans_kernel<<<g, 256, 0, stream>>>(wx2, wx2T, RNN_H, RNN_H, RNN_H);
    wtrans_kernel<<<g, 256, 0, stream>>>(wh2, wh2T, RNN_H, RNN_H, RNN_H);
  }

  // embedding GEMM: [B*T, 300] @ [300, 512] -> bf16 [T][B][512]
  emb_kernel<<<(RNN_B * RNN_T) / MTILE, 256, 0, stream>>>(x, WembT, b_emb, emb);

  // recurrence + final FC
  rnn_kernel<<<RNN_B / MTILE, 256, 0, stream>>>(emb,
      wx0T, wh0T, b0, wx1T, wh1T, b1, wx2T, wh2T, b2, W_fc, b_fc, out);
}
